// RNN_1709396984256
// MI455X (gfx1250) — compile-verified, matrix operand order checked by reading, not run on hardware
//
#include <hip/hip_runtime.h>
#include <hip/hip_bf16.h>
#include <cstdint>

// ---------------------------------------------------------------------------
// 2-layer LSTM + FC for MI455X (gfx1250), wave32, bf16 WMMA w/ f32 accumulate.
// Recurrence runs as ONE persistent kernel per layer (grid-wide software
// barrier between timesteps) to kill per-step launch overhead.
// ---------------------------------------------------------------------------

typedef __bf16 v8bf  __attribute__((ext_vector_type(8)));
typedef __bf16 v16bf __attribute__((ext_vector_type(16)));
typedef float  v8f   __attribute__((ext_vector_type(8)));

static constexpr int Bsz = 64, Tsz = 256, DIN = 512, Hsz = 1024, Csz = 1000;
static constexpr int G4H = 4 * Hsz; // 4096
static constexpr int HC  = 32;      // h-columns per workgroup in recurrent kernel

static __device__ __forceinline__ unsigned short f32_to_bf16_rne(float f) {
  unsigned int u = __float_as_uint(f);
  if ((u & 0x7fffffffu) > 0x7f800000u) return (unsigned short)((u >> 16) | 0x0040u); // quiet NaN
  u += 0x7fffu + ((u >> 16) & 1u);
  return (unsigned short)(u >> 16);
}

// Fragment layout for bf16 16x32 "A" (and 32x16 "B" stored as [N x K] row-major,
// i.e. B[k][n] = W[n][k] so a W row supplies one B column):
//   lanes 0-15 : row r0+lane,    K = {k0..k0+7,   k0+16..k0+23}
//   lanes16-31 : row r0+lane-16, K = {k0+8..k0+15, k0+24..k0+31}
// Two contiguous 16-byte loads per lane.
static __device__ __forceinline__ v16bf load_frag_bf16(
    const unsigned short* __restrict__ base, int ld, int r0, int k0, int lane) {
  int row  = lane & 15;
  int half = (lane >> 4) & 1;
  const unsigned short* p = base + (size_t)(r0 + row) * (size_t)ld + k0 + half * 8;
  v8bf lo = *(const v8bf*)(p);
  v8bf hi = *(const v8bf*)(p + 16);
  return __builtin_shufflevector(lo, hi, 0,1,2,3,4,5,6,7,8,9,10,11,12,13,14,15);
}

static __device__ __forceinline__ v8f wmma_bf16(v16bf a, v16bf b, v8f c) {
  // (neg_a, A, neg_b, B, c_mod, C, reuse_a, reuse_b)
  return __builtin_amdgcn_wmma_f32_16x16x32_bf16(false, a, false, b, (short)0, c, false, false);
}

// ---------------------------------------------------------------------------
// Utility kernels
// ---------------------------------------------------------------------------
__global__ void cvt_bf16_kernel(const float* __restrict__ src, unsigned short* __restrict__ dst,
                                long long n_src, long long n_dst) {
  long long i = (long long)blockIdx.x * blockDim.x + threadIdx.x;
  long long stride = (long long)gridDim.x * blockDim.x;
  for (; i < n_dst; i += stride)
    dst[i] = (i < n_src) ? f32_to_bf16_rne(src[i]) : (unsigned short)0;
}

__global__ void bias_sum_kernel(const float* __restrict__ a, const float* __restrict__ b,
                                float* __restrict__ out, int n) {
  int i = blockIdx.x * blockDim.x + threadIdx.x;
  if (i < n) out[i] = a[i] + b[i];
}

__global__ void init_state_kernel(unsigned short* __restrict__ h, float* __restrict__ c,
                                  unsigned int* __restrict__ ctr, int n) {
  int i = blockIdx.x * blockDim.x + threadIdx.x;
  if (i < n) { h[i] = 0; c[i] = 0.0f; }
  if (i == 0) *ctr = 0u;
}

// ---------------------------------------------------------------------------
// Batched GEMM: out[M,N] f32 = A[M,K] bf16  x  W[N,K]^T bf16  + bias[N]
// 256 threads (8 waves); block tile 128(M) x 64(N); grid(x = N/64, y = M/128).
// Wave w covers rows [m0 + 16w, +16) against all 64 block columns.
// ---------------------------------------------------------------------------
__global__ __launch_bounds__(256) void gemm_bias_kernel(
    const unsigned short* __restrict__ A, const unsigned short* __restrict__ W,
    const float* __restrict__ bias, float* __restrict__ out,
    int M, int N, int K) {
  int wave = threadIdx.x >> 5;
  int lane = threadIdx.x & 31;
  int m0 = blockIdx.y * 128 + wave * 16;
  int n0 = blockIdx.x * 64;
  v8f acc[4] = {};
  for (int k0 = 0; k0 < K; k0 += 32) {
    v16bf a = load_frag_bf16(A, K, m0, k0, lane);
#pragma unroll
    for (int j = 0; j < 4; ++j) {
      v16bf b = load_frag_bf16(W, K, n0 + j * 16, k0, lane);
      acc[j] = wmma_bf16(a, b, acc[j]);
    }
  }
  int row = lane & 15, half = (lane >> 4) & 1;
#pragma unroll
  for (int j = 0; j < 4; ++j) {
    int n = n0 + j * 16 + row;
    float bn = bias[n];
#pragma unroll
    for (int r = 0; r < 8; ++r) {
      int m = m0 + half * 8 + r;   // C/D layout: lanes16-31 hold M=8..15
      out[(size_t)m * (size_t)N + n] = acc[j][r] + bn;
    }
  }
}

// ---------------------------------------------------------------------------
// Persistent LSTM layer kernel: loops all T timesteps internally.
// grid = H/HC = 32 blocks, 256 threads (8 waves). Each block owns a 32-wide
// h-column slice: per step it computes the 4 gate tiles (64xHC) for its slice
// via WMMA into LDS, applies the nonlinearities / c,h update, then passes a
// grid-wide software barrier (release fence -> atomic counter -> acquire
// fence, the acquire invalidating WGP$ so next step's h loads are fresh).
// h is ping-pong buffered by timestep parity; final h lands in hA (T even).
// ---------------------------------------------------------------------------
__global__ __launch_bounds__(256) void lstm_layer_kernel(
    const float* __restrict__ xg,            // [B*T, 4H], row = b*T + t
    const unsigned short* __restrict__ Whh,  // [4H, H] bf16
    unsigned short* hA,                      // [B, H] bf16 (zero-initialized)
    unsigned short* hB,                      // [B, H] bf16
    float* __restrict__ c,                   // [B, H] f32 (zero-initialized)
    unsigned short* __restrict__ seq,        // [B, T, H] bf16 or nullptr
    unsigned int* __restrict__ ctr,          // zero-initialized barrier counter
    int store_seq) {
  __shared__ float gl[4][64][HC];            // 32 KB gate staging
  int wave = threadIdx.x >> 5;
  int lane = threadIdx.x & 31;
  int hc0  = blockIdx.x * HC;
  int mt   = wave & 3;            // M-tile 0..3 (M = B = 64)
  int p    = wave >> 2;           // waves 0-3: gates {0,1}; waves 4-7: {2,3}
  int row  = lane & 15, half = (lane >> 4) & 1;
  unsigned int nb = gridDim.x;

  for (int t = 0; t < Tsz; ++t) {
    const unsigned short* h_in = (t & 1) ? hB : hA;
    unsigned short* h_out      = (t & 1) ? hA : hB;

    v8f acc[4] = {};
    for (int k0 = 0; k0 < Hsz; k0 += 32) {
      v16bf a = load_frag_bf16(h_in, Hsz, mt * 16, k0, lane);  // A reused 4x
#pragma unroll
      for (int j = 0; j < 4; ++j) {
        int gn = p * 4 + j, g = gn >> 1, nt = gn & 1;
        v16bf b = load_frag_bf16(Whh, Hsz, g * Hsz + hc0 + nt * 16, k0, lane);
        acc[j] = wmma_bf16(a, b, acc[j]);
      }
    }
#pragma unroll
    for (int j = 0; j < 4; ++j) {
      int gn = p * 4 + j, g = gn >> 1, nt = gn & 1;
#pragma unroll
      for (int r = 0; r < 8; ++r)
        gl[g][mt * 16 + half * 8 + r][nt * 16 + row] = acc[j][r];
    }
    __syncthreads();

#pragma unroll
    for (int it = 0; it < 8; ++it) {
      int idx = threadIdx.x + it * 256;   // 64*HC elements / 256 threads
      int m  = idx >> 5;
      int hl = idx & (HC - 1);
      size_t xrow = ((size_t)m * Tsz + t) * (size_t)G4H + hc0 + hl;
      float iv = gl[0][m][hl] + xg[xrow];
      float fv = gl[1][m][hl] + xg[xrow + (size_t)Hsz];
      float gv = gl[2][m][hl] + xg[xrow + (size_t)(2 * Hsz)];
      float ov = gl[3][m][hl] + xg[xrow + (size_t)(3 * Hsz)];
      iv = 1.0f / (1.0f + __expf(-iv));
      fv = 1.0f / (1.0f + __expf(-fv));
      gv = tanhf(gv);
      ov = 1.0f / (1.0f + __expf(-ov));
      int hidx = m * Hsz + hc0 + hl;
      float cv = fv * c[hidx] + iv * gv;
      c[hidx] = cv;
      float hv = ov * tanhf(cv);
      unsigned short hb = f32_to_bf16_rne(hv);
      h_out[hidx] = hb;
      if (store_seq) seq[((size_t)m * Tsz + t) * (size_t)Hsz + hc0 + hl] = hb;
    }

    // ---- grid-wide barrier between timesteps ----
    __builtin_amdgcn_fence(__ATOMIC_RELEASE, "agent");   // drain our h stores to L2
    __syncthreads();
    if (threadIdx.x == 0) {
      __hip_atomic_fetch_add(ctr, 1u, __ATOMIC_RELAXED, __HIP_MEMORY_SCOPE_AGENT);
      unsigned int target = (unsigned int)(t + 1) * nb;  // monotonic, no reset race
      while (__hip_atomic_load(ctr, __ATOMIC_RELAXED, __HIP_MEMORY_SCOPE_AGENT) < target)
        __builtin_amdgcn_s_sleep(2);
    }
    __syncthreads();
    __builtin_amdgcn_fence(__ATOMIC_ACQUIRE, "agent");   // invalidate WGP$ before h reads
  }
}

// ---------------------------------------------------------------------------
// FC: out[B, C] = h[B,H] bf16 @ fcw[Npad=1024, H]^T bf16 + fcb.  grid = 8 WGs.
// ---------------------------------------------------------------------------
__global__ __launch_bounds__(256) void fc_kernel(
    const unsigned short* __restrict__ h, const unsigned short* __restrict__ W,
    const float* __restrict__ bias, float* __restrict__ out) {
  int wave = threadIdx.x >> 5;
  int lane = threadIdx.x & 31;
  int n0 = (blockIdx.x * 8 + wave) * 16;
  v8f acc[4] = {};
  for (int k0 = 0; k0 < Hsz; k0 += 32) {
    v16bf b = load_frag_bf16(W, Hsz, n0, k0, lane);
#pragma unroll
    for (int mt = 0; mt < 4; ++mt) {
      v16bf a = load_frag_bf16(h, Hsz, mt * 16, k0, lane);
      acc[mt] = wmma_bf16(a, b, acc[mt]);
    }
  }
  int row = lane & 15, half = (lane >> 4) & 1;
  int n = n0 + row;
  if (n < Csz) {
    float bn = bias[n];
#pragma unroll
    for (int mt = 0; mt < 4; ++mt)
#pragma unroll
      for (int r = 0; r < 8; ++r) {
        int m = mt * 16 + half * 8 + r;
        out[(size_t)m * Csz + n] = acc[mt][r] + bn;
      }
  }
}

// ---------------------------------------------------------------------------
// Host launcher
// ---------------------------------------------------------------------------
extern "C" void kernel_launch(void* const* d_in, const int* in_sizes, int n_in,
                              void* d_out, int out_size, void* d_ws, size_t ws_size,
                              hipStream_t stream) {
  const float* x    = (const float*)d_in[0];
  const float* Wih0 = (const float*)d_in[1];
  const float* Whh0 = (const float*)d_in[2];
  const float* bih0 = (const float*)d_in[3];
  const float* bhh0 = (const float*)d_in[4];
  const float* Wih1 = (const float*)d_in[5];
  const float* Whh1 = (const float*)d_in[6];
  const float* bih1 = (const float*)d_in[7];
  const float* bhh1 = (const float*)d_in[8];
  const float* fcw  = (const float*)d_in[9];
  const float* fcb  = (const float*)d_in[10];
  float* out = (float*)d_out;

  uint8_t* ws = (uint8_t*)d_ws;
  size_t off = 0;
  auto alloc = [&](size_t bytes) -> void* {
    void* p = ws + off;
    off += (bytes + 255) & ~(size_t)255;
    return p;
  };

  unsigned short* xbf   = (unsigned short*)alloc((size_t)Bsz * Tsz * DIN * 2);
  unsigned short* wih0b = (unsigned short*)alloc((size_t)G4H * DIN * 2);
  unsigned short* whh0b = (unsigned short*)alloc((size_t)G4H * Hsz * 2);
  unsigned short* wih1b = (unsigned short*)alloc((size_t)G4H * Hsz * 2);
  unsigned short* whh1b = (unsigned short*)alloc((size_t)G4H * Hsz * 2);
  unsigned short* fcwb  = (unsigned short*)alloc((size_t)1024 * Hsz * 2);  // zero-padded 1000->1024
  float* bias0          = (float*)alloc((size_t)G4H * 4);
  float* bias1          = (float*)alloc((size_t)G4H * 4);
  float* xg             = (float*)alloc((size_t)Bsz * Tsz * G4H * 4);     // reused by both layers
  unsigned short* seq   = (unsigned short*)alloc((size_t)Bsz * Tsz * Hsz * 2);
  unsigned short* hA    = (unsigned short*)alloc((size_t)Bsz * Hsz * 2);
  unsigned short* hB    = (unsigned short*)alloc((size_t)Bsz * Hsz * 2);
  float* cst            = (float*)alloc((size_t)Bsz * Hsz * 4);
  unsigned int* ctr     = (unsigned int*)alloc(256);
  (void)ws_size; (void)in_sizes; (void)n_in; (void)out_size;

  auto cvt = [&](const float* s, unsigned short* d, long long nsrc, long long ndst) {
    cvt_bf16_kernel<<<2048, 256, 0, stream>>>(s, d, nsrc, ndst);
  };

  // 1) precision conversion + bias pre-sums
  cvt(x,    xbf,   (long long)Bsz * Tsz * DIN, (long long)Bsz * Tsz * DIN);
  cvt(Wih0, wih0b, (long long)G4H * DIN,       (long long)G4H * DIN);
  cvt(Whh0, whh0b, (long long)G4H * Hsz,       (long long)G4H * Hsz);
  cvt(Wih1, wih1b, (long long)G4H * Hsz,       (long long)G4H * Hsz);
  cvt(Whh1, whh1b, (long long)G4H * Hsz,       (long long)G4H * Hsz);
  cvt(fcw,  fcwb,  (long long)Csz * Hsz,       (long long)1024 * Hsz);
  bias_sum_kernel<<<G4H / 256, 256, 0, stream>>>(bih0, bhh0, bias0, G4H);
  bias_sum_kernel<<<G4H / 256, 256, 0, stream>>>(bih1, bhh1, bias1, G4H);

  // 2) layer-0 input GEMM: xg = x @ Wih0^T + (b_ih0 + b_hh0)
  {
    dim3 g(G4H / 64, (Bsz * Tsz) / 128);
    gemm_bias_kernel<<<g, 256, 0, stream>>>(xbf, wih0b, bias0, xg, Bsz * Tsz, G4H, DIN);
  }

  // 3) layer-0 recurrence: ONE persistent kernel over all 256 steps
  init_state_kernel<<<(Bsz * Hsz) / 256, 256, 0, stream>>>(hA, cst, ctr, Bsz * Hsz);
  lstm_layer_kernel<<<Hsz / HC, 256, 0, stream>>>(xg, whh0b, hA, hB, cst, seq, ctr, 1);

  // 4) layer-1 input GEMM: xg = h1seq @ Wih1^T + (b_ih1 + b_hh1)
  {
    dim3 g(G4H / 64, (Bsz * Tsz) / 128);
    gemm_bias_kernel<<<g, 256, 0, stream>>>(seq, wih1b, bias1, xg, Bsz * Tsz, G4H, Hsz);
  }

  // 5) layer-1 recurrence (no sequence store; final h lands in hA since T is even)
  init_state_kernel<<<(Bsz * Hsz) / 256, 256, 0, stream>>>(hA, cst, ctr, Bsz * Hsz);
  lstm_layer_kernel<<<Hsz / HC, 256, 0, stream>>>(xg, whh1b, hA, hB, cst, seq, ctr, 0);

  // 6) FC on last hidden state
  fc_kernel<<<8, 256, 0, stream>>>(hA, fcwb, fcb, out);
}
